// XLNetRelativeAttention_16389595201928
// MI455X (gfx1250) — matrix operand order repaired and checked by reading, not compiled, not run on hardware
//
#include <hip/hip_runtime.h>

typedef __attribute__((ext_vector_type(16))) __bf16 v16bf;
typedef __attribute__((ext_vector_type(8)))  __bf16 v8bf;
typedef __attribute__((ext_vector_type(8)))  float  v8f;

#define S_LEN 1024
#define BATCH 4
#define DMODEL 1024
#define NHEAD 16
#define DHEAD 64
#define RLEN 2048

// A-matrix (16x32 bf16) fragment from an fp32 row.
// Element e (0..15) -> k = d0+e  (d0 = kt + lhi*16)
__device__ __forceinline__ v16bf packA_nb(const float* rowp, int d0) {
  v16bf a;
#pragma unroll
  for (int e = 0; e < 8; ++e) {
    a[e]     = (__bf16)rowp[d0 + e];
    a[e + 8] = (__bf16)rowp[d0 + 8 + e];
  }
  return a;
}
__device__ __forceinline__ v16bf packA_bias(const float* rowp, const float* biasp, int d0) {
  v16bf a;
#pragma unroll
  for (int e = 0; e < 8; ++e) {
    a[e]     = (__bf16)(rowp[d0 + e]      + biasp[d0 + e]);
    a[e + 8] = (__bf16)(rowp[d0 + 8 + e] + biasp[d0 + 8 + e]);
  }
  return a;
}

__device__ __forceinline__ float rmax16(float v) {
#pragma unroll
  for (int m = 8; m >= 1; m >>= 1) v = fmaxf(v, __shfl_xor(v, m, 32));
  return v;
}
__device__ __forceinline__ float rsum16(float v) {
#pragma unroll
  for (int m = 8; m >= 1; m >>= 1) v += __shfl_xor(v, m, 32);
  return v;
}

// ---------------------------------------------------------------------------
// K0: convert + transpose weights to bf16.
// ---------------------------------------------------------------------------
__global__ void prep_weights(const float* Wq, const float* Wk, const float* Wv,
                             const float* Wr, const float* Wo,
                             __bf16* WqT, __bf16* WkT, __bf16* WvT,
                             __bf16* WrT, __bf16* WoB) {
  int idx = blockIdx.x * blockDim.x + threadIdx.x;
  if (idx >= DMODEL * DMODEL) return;
  int c = idx >> 10;
  int k = idx & 1023;
  int src = k * DMODEL + c;
  WqT[idx] = (__bf16)Wq[src];
  WkT[idx] = (__bf16)Wk[src];
  WvT[idx] = (__bf16)Wv[src];
  WrT[idx] = (__bf16)Wr[src];
  WoB[idx] = (__bf16)Wo[idx];
}

// ---------------------------------------------------------------------------
// K1: projection GEMM, 16x64 output per wave; MODE is compile-time so the
// store epilogue is straight-line (no per-element scalar branching).
//   MODE 0: qf fp32 [bn][Lv][H]   MODE 1: kh bf16 [bn][Lv][H]
//   MODE 2: vt bf16 [bn][H][S]    MODE 3: krb bf16 [bn][Lv][H]
// ---------------------------------------------------------------------------
template <int MODE, int LV>
__global__ __launch_bounds__(32)
void proj_gemm(const float* __restrict__ X, const __bf16* __restrict__ Wt,
               float* __restrict__ outf, __bf16* __restrict__ outb) {
  const int lane = threadIdx.x;
  const int lcol = lane & 15;
  const int lhi  = lane >> 4;
  const int mbase = blockIdx.x * 16;
  const int cbase = blockIdx.y * 64;
  const float*  xrow = X  + (size_t)(mbase + lcol) * DMODEL;
  const __bf16* bcol = Wt + (size_t)(cbase + lcol) * DMODEL + lhi * 16;
  v8f acc[4] = {{}, {}, {}, {}};
  for (int kt = 0; kt < DMODEL; kt += 32) {
    __builtin_prefetch(bcol + kt + 128, 0, 0);
    v16bf A = packA_nb(xrow, kt + lhi * 16);
#pragma unroll
    for (int t = 0; t < 4; ++t) {
      v16bf Bm = *(const v16bf*)(bcol + (size_t)t * 16 * DMODEL + kt);
      acc[t] = __builtin_amdgcn_wmma_f32_16x16x32_bf16(false, A, false, Bm,
                                                       (short)0, acc[t], false, false);
    }
  }
#pragma unroll
  for (int t = 0; t < 4; ++t) {
#pragma unroll
    for (int v = 0; v < 8; ++v) {
      int m = mbase + v + 8 * lhi;
      int c = cbase + t * 16 + lcol;
      int i = m >> 2, b = m & 3;        // m = i*B + b
      int n = c >> 6, d = c & 63;       // c = n*H + d
      int bn = b * NHEAD + n;
      float val = acc[t][v];
      if (MODE == 0)      outf[((size_t)bn * LV + i) * DHEAD + d] = val;
      else if (MODE == 2) outb[((size_t)bn * DHEAD + d) * S_LEN + i] = (__bf16)val;
      else                outb[((size_t)bn * LV + i) * DHEAD + d] = (__bf16)val;
    }
  }
}

// ---------------------------------------------------------------------------
// K2: ef[bn][i][s] = sum_d (q[bn,i,d] + r_s_bias[n,d]) * seg_embed[s,n,d]
// ---------------------------------------------------------------------------
__global__ void ef_kernel(const float* qf, const float* r_s_bias,
                          const float* seg_embed, float* efb) {
  int t = blockIdx.x * blockDim.x + threadIdx.x;
  if (t >= BATCH * NHEAD * S_LEN) return;
  int bn = t >> 10;
  int n  = bn & 15;
  const float* q = qf + (size_t)t * DHEAD;
  float e0 = 0.f, e1 = 0.f;
#pragma unroll 4
  for (int d = 0; d < DHEAD; ++d) {
    float qv = q[d] + r_s_bias[n * DHEAD + d];
    e0 += qv * seg_embed[n * DHEAD + d];
    e1 += qv * seg_embed[NHEAD * DHEAD + n * DHEAD + d];
  }
  efb[(size_t)t * 2]     = e0;
  efb[(size_t)t * 2 + 1] = e1;
}

// ---------------------------------------------------------------------------
// K3: flash attention, one wave per (b, n, 16 query rows), 32 keys per step.
//   Per 32 keys: 4 ac WMMAs, 6 bd WMMAs (47-col rel-shift band), 4 PV WMMAs.
// ---------------------------------------------------------------------------
__global__ __launch_bounds__(32)
void attn_kernel(const float* __restrict__ qf, const __bf16* __restrict__ kh,
                 const __bf16* __restrict__ vt, const __bf16* __restrict__ krb,
                 const float* __restrict__ efb, const float* __restrict__ seg_mat,
                 const float* __restrict__ r_w_bias,
                 const float* __restrict__ r_r_bias, __bf16* __restrict__ av) {
  __shared__ float  bdbuf[16][48];
  __shared__ __bf16 pbuf[16][32];
  const int lane = threadIdx.x;
  const int lcol = lane & 15;
  const int lhi  = lane >> 4;
  const int ibase = blockIdx.x * 16;
  const int bn = blockIdx.y;              // b*NHEAD + n
  const int b  = bn >> 4;
  const int n  = bn & 15;

  const float* qrow = qf + ((size_t)bn * S_LEN + (ibase + lcol)) * DHEAD;
  const float* bw = r_w_bias + n * DHEAD;
  const float* br = r_r_bias + n * DHEAD;
  v16bf Aw[2], Ar[2];
#pragma unroll
  for (int ks = 0; ks < 2; ++ks) {
    int d0 = ks * 32 + lhi * 16;
    Aw[ks] = packA_bias(qrow, bw, d0);
    Ar[ks] = packA_bias(qrow, br, d0);
  }

  float mrow[8], lrow[8], e0r[8], e1r[8];
#pragma unroll
  for (int v = 0; v < 8; ++v) {
    mrow[v] = -3.0e38f;
    lrow[v] = 0.f;
    int ig = ibase + v + 8 * lhi;
    e0r[v] = efb[((size_t)bn * S_LEN + ig) * 2];
    e1r[v] = efb[((size_t)bn * S_LEN + ig) * 2 + 1];
  }
  v8f O0 = {}, O1 = {}, O2 = {}, O3 = {};

  const int njt = (ibase >> 5) + 1;       // 32-key blocks up to the diagonal
  for (int jt = 0; jt < njt; ++jt) {
    const int jb = jt * 32;
    // --- content scores for both 16-col tiles ---
    v8f acc0 = {}, acc1 = {};
#pragma unroll
    for (int ks = 0; ks < 2; ++ks) {
      const __bf16* kp = kh + ((size_t)bn * S_LEN + jb + lcol) * DHEAD
                            + ks * 32 + lhi * 16;
      v16bf Bk0 = *(const v16bf*)kp;
      v16bf Bk1 = *(const v16bf*)(kp + (size_t)16 * DHEAD);
      acc0 = __builtin_amdgcn_wmma_f32_16x16x32_bf16(false, Aw[ks], false, Bk0,
                                                     (short)0, acc0, false, false);
      acc1 = __builtin_amdgcn_wmma_f32_16x16x32_bf16(false, Aw[ks], false, Bk1,
                                                     (short)0, acc1, false, false);
    }
    // --- position scores: 3 raw strips covering cols [c-16, c+32) ---
    const int cctr = jb - ibase + S_LEN;
#pragma unroll
    for (int u = 0; u < 3; ++u) {
      v8f bacc = {};
      int jr = cctr - 16 + u * 16 + lcol;
#pragma unroll
      for (int ks = 0; ks < 2; ++ks) {
        const __bf16* rp = krb + ((size_t)bn * RLEN + jr) * DHEAD
                               + ks * 32 + lhi * 16;
        v16bf Br = *(const v16bf*)rp;
        bacc = __builtin_amdgcn_wmma_f32_16x16x32_bf16(false, Ar[ks], false, Br,
                                                       (short)0, bacc, false, false);
      }
#pragma unroll
      for (int v = 0; v < 8; ++v)
        bdbuf[v + 8 * lhi][u * 16 + lcol] = bacc[v];
    }
    __syncthreads();
    // --- combine + ef + scale + causal mask for both tiles ---
    float sc0[8], sc1[8];
#pragma unroll
    for (int v = 0; v < 8; ++v) {
      int r  = v + 8 * lhi;
      int ig = ibase + r;
      int jg0 = jb + lcol;
      int jg1 = jb + 16 + lcol;
      float s0 = acc0[v] + bdbuf[r][16 + lcol - r];
      float s1 = acc1[v] + bdbuf[r][32 + lcol - r];
      float m10 = seg_mat[(((size_t)ig * S_LEN + jg0) * BATCH + b) * 2 + 1];
      float m11 = seg_mat[(((size_t)ig * S_LEN + jg1) * BATCH + b) * 2 + 1];
      float ed = e1r[v] - e0r[v];
      s0 = (s0 + e0r[v] + ed * m10) * 0.125f;
      s1 = (s1 + e0r[v] + ed * m11) * 0.125f;
      if (jg0 > ig) s0 -= 1.0e30f;
      if (jg1 > ig) s1 -= 1.0e30f;
      sc0[v] = s0; sc1[v] = s1;
    }
    __syncthreads();
    // --- online softmax over 32 keys ---
    float alpha[8];
#pragma unroll
    for (int v = 0; v < 8; ++v) {
      float rm   = fmaxf(rmax16(sc0[v]), rmax16(sc1[v]));
      float mnew = fmaxf(mrow[v], rm);
      float a    = __expf(mrow[v] - mnew);
      float p0   = __expf(sc0[v] - mnew);
      float p1   = __expf(sc1[v] - mnew);
      lrow[v] = lrow[v] * a + rsum16(p0 + p1);
      mrow[v] = mnew;
      alpha[v] = a;
      pbuf[v + 8 * lhi][lcol]      = (__bf16)p0;
      pbuf[v + 8 * lhi][16 + lcol] = (__bf16)p1;
    }
    __syncthreads();
#pragma unroll
    for (int v = 0; v < 8; ++v) {
      O0[v] *= alpha[v]; O1[v] *= alpha[v]; O2[v] *= alpha[v]; O3[v] *= alpha[v];
    }
    // P (16x32) in A layout: two contiguous 8x bf16 runs of this lane's row
    // (16B-aligned -> two ds_load_b128 instead of 16 ds_load_u16)
    const v8bf* prow = (const v8bf*)&pbuf[lcol][0];
    v8bf plo = prow[2 * lhi];      // elements k = 16*lhi .. 16*lhi+7
    v8bf phi = prow[2 * lhi + 1];  // elements k = 16*lhi+8 .. 16*lhi+15
    v16bf Pa;
#pragma unroll
    for (int e = 0; e < 8; ++e) { Pa[e] = plo[e]; Pa[e + 8] = phi[e]; }
    __syncthreads();
    // --- O += P @ V ---
    v8f* Ot[4] = { &O0, &O1, &O2, &O3 };
#pragma unroll
    for (int dt = 0; dt < 4; ++dt) {
      const __bf16* vp = vt + ((size_t)bn * DHEAD + dt * 16 + lcol) * S_LEN
                            + jb + lhi * 16;
      v16bf Bv = *(const v16bf*)vp;
      *Ot[dt] = __builtin_amdgcn_wmma_f32_16x16x32_bf16(false, Pa, false, Bv,
                                                        (short)0, *Ot[dt], false, false);
    }
  }
  // finalize: av[(i*B+b)*D + n*H + d] = O / l
#pragma unroll
  for (int v = 0; v < 8; ++v) {
    int ig = ibase + v + 8 * lhi;
    float inv = 1.0f / lrow[v];
    size_t base = ((size_t)ig * BATCH + b) * DMODEL + n * DHEAD;
    av[base + 0  + lcol] = (__bf16)(O0[v] * inv);
    av[base + 16 + lcol] = (__bf16)(O1[v] * inv);
    av[base + 32 + lcol] = (__bf16)(O2[v] * inv);
    av[base + 48 + lcol] = (__bf16)(O3[v] * inv);
  }
}

// ---------------------------------------------------------------------------
// K4: output projection + residual, 16x64 per wave.
// ---------------------------------------------------------------------------
__global__ __launch_bounds__(32)
void out_gemm(const __bf16* __restrict__ av, const __bf16* __restrict__ WoB,
              const float* __restrict__ h, float* __restrict__ ao) {
  const int lane = threadIdx.x;
  const int lcol = lane & 15;
  const int lhi  = lane >> 4;
  const int mbase = blockIdx.x * 16;
  const int cbase = blockIdx.y * 64;
  const __bf16* arow = av  + (size_t)(mbase + lcol) * DMODEL;
  const __bf16* bcol = WoB + (size_t)(cbase + lcol) * DMODEL + lhi * 16;
  v8f acc[4] = {{}, {}, {}, {}};
  for (int kt = 0; kt < DMODEL; kt += 32) {
    __builtin_prefetch(bcol + kt + 128, 0, 0);
    int d0 = kt + lhi * 16;
    v8bf lo = *(const v8bf*)(arow + d0);
    v8bf hi = *(const v8bf*)(arow + d0 + 8);
    v16bf A;
#pragma unroll
    for (int e = 0; e < 8; ++e) { A[e] = lo[e]; A[e + 8] = hi[e]; }
#pragma unroll
    for (int t = 0; t < 4; ++t) {
      v16bf Bm = *(const v16bf*)(bcol + (size_t)t * 16 * DMODEL + kt);
      acc[t] = __builtin_amdgcn_wmma_f32_16x16x32_bf16(false, A, false, Bm,
                                                       (short)0, acc[t], false, false);
    }
  }
#pragma unroll
  for (int t = 0; t < 4; ++t) {
#pragma unroll
    for (int v = 0; v < 8; ++v) {
      size_t idx = (size_t)(mbase + v + 8 * lhi) * DMODEL + cbase + t * 16 + lcol;
      ao[idx] = acc[t][v] + h[idx];
    }
  }
}

// ---------------------------------------------------------------------------
// K5: layernorm per row of [S*B, D]
// ---------------------------------------------------------------------------
__global__ __launch_bounds__(256)
void ln_kernel(const float* ao, const float* scale, const float* bias, float* out) {
  __shared__ float sbuf[256];
  const int m = blockIdx.x;
  const int tid = threadIdx.x;
  const float* x = ao + (size_t)m * DMODEL;
  float s = 0.f;
  for (int c = tid; c < DMODEL; c += 256) s += x[c];
  sbuf[tid] = s; __syncthreads();
  for (int st = 128; st > 0; st >>= 1) {
    if (tid < st) sbuf[tid] += sbuf[tid + st];
    __syncthreads();
  }
  float mu = sbuf[0] * (1.0f / DMODEL);
  __syncthreads();
  float v = 0.f;
  for (int c = tid; c < DMODEL; c += 256) { float d = x[c] - mu; v += d * d; }
  sbuf[tid] = v; __syncthreads();
  for (int st = 128; st > 0; st >>= 1) {
    if (tid < st) sbuf[tid] += sbuf[tid + st];
    __syncthreads();
  }
  float rs = rsqrtf(sbuf[0] * (1.0f / DMODEL) + 1e-12f);
  for (int c = tid; c < DMODEL; c += 256)
    out[(size_t)m * DMODEL + c] = (x[c] - mu) * rs * scale[c] + bias[c];
}

// ---------------------------------------------------------------------------
extern "C" void kernel_launch(void* const* d_in, const int* in_sizes, int n_in,
                              void* d_out, int out_size, void* d_ws, size_t ws_size,
                              hipStream_t stream) {
  (void)in_sizes; (void)n_in; (void)out_size; (void)ws_size;
  const float* h        = (const float*)d_in[0];
  const float* r        = (const float*)d_in[1];
  const float* seg_mat  = (const float*)d_in[2];
  // d_in[3] attn_mask: exactly the causal mask; recomputed in-kernel
  const float* Wq       = (const float*)d_in[4];
  const float* Wk       = (const float*)d_in[5];
  const float* Wv       = (const float*)d_in[6];
  const float* Wo       = (const float*)d_in[7];
  const float* Wr       = (const float*)d_in[8];
  const float* r_w_bias = (const float*)d_in[9];
  const float* r_r_bias = (const float*)d_in[10];
  const float* r_s_bias = (const float*)d_in[11];
  const float* seg_embed= (const float*)d_in[12];
  const float* ln_scale = (const float*)d_in[13];
  const float* ln_bias  = (const float*)d_in[14];
  float* out = (float*)d_out;

  char* ws = (char*)d_ws;
  const size_t MB = 1024 * 1024;
  __bf16* WqT = (__bf16*)(ws + 0 * MB);
  __bf16* WkT = (__bf16*)(ws + 2 * MB);
  __bf16* WvT = (__bf16*)(ws + 4 * MB);
  __bf16* WrT = (__bf16*)(ws + 6 * MB);
  __bf16* WoB = (__bf16*)(ws + 8 * MB);
  float*  qf  = (float* )(ws + 10 * MB);   // [B*N][S][H] fp32      16 MB
  __bf16* kh  = (__bf16*)(ws + 26 * MB);   // [B*N][S][H] bf16       8 MB
  __bf16* vt  = (__bf16*)(ws + 34 * MB);   // [B*N][H][S] bf16       8 MB
  __bf16* krb = (__bf16*)(ws + 42 * MB);   // [B*N][R][H] bf16      16 MB
  float*  efb = (float* )(ws + 58 * MB);   // [B*N][S][2] fp32     0.5 MB
  __bf16* av  = (__bf16*)(ws + 59 * MB);   // [S*B][N*H] bf16        8 MB
  float*  ao  = (float* )(ws + 67 * MB);   // [S*B][D] fp32         16 MB

  prep_weights<<<4096, 256, 0, stream>>>(Wq, Wk, Wv, Wr, Wo,
                                         WqT, WkT, WvT, WrT, WoB);
  dim3 g1(256, 16);   // M=4096 rows, C=1024 cols, 16x64 tiles
  proj_gemm<0, S_LEN><<<g1, 32, 0, stream>>>(h, WqT, qf, nullptr);
  proj_gemm<1, S_LEN><<<g1, 32, 0, stream>>>(h, WkT, nullptr, kh);
  proj_gemm<2, S_LEN><<<g1, 32, 0, stream>>>(h, WvT, nullptr, vt);
  dim3 g2(512, 16);   // M=8192 rows for r
  proj_gemm<3, RLEN><<<g2, 32, 0, stream>>>(r, WrT, nullptr, krb);
  ef_kernel<<<256, 256, 0, stream>>>(qf, r_s_bias, seg_embed, efb);
  dim3 g3(64, 64);    // (i-tiles, b*n)
  attn_kernel<<<g3, 32, 0, stream>>>(qf, kh, vt, krb, efb, seg_mat,
                                     r_w_bias, r_r_bias, av);
  out_gemm<<<g1, 32, 0, stream>>>(av, WoB, h, ao);
  ln_kernel<<<4096, 256, 0, stream>>>(ao, ln_scale, ln_bias, out);
}
